// ReweightSample_24962349924822
// MI455X (gfx1250) — matrix operand order repaired and checked by reading, not compiled, not run on hardware
//
#include <hip/hip_runtime.h>
#include <hip/hip_bf16.h>

// ---------------------------------------------------------------------------
// ReweightSample for MI455X (gfx1250, wave32, WMMA f16 16x16x32 + async->LDS)
//
// out[i,j] = exp( (L[i,j] - rowmax_i(L)) / 20 ),
//   L = (k kT) / (g gT + 1e-7),  diag(L) = -10,
//   k,g = row-L2-normalized feat_k, feat_g (x / (||x|| + 1e-7)).
// (softmax sum cancels against the row-max renormalization.)
//
// GEMM: 128x128 workgroup tile, B panel staged via GLOBAL_LOAD_ASYNC_TO_LDS
// (double-buffered 32-K slices, ASYNCcnt-pipelined), A per-wave from global.
// ---------------------------------------------------------------------------

typedef __attribute__((ext_vector_type(16))) _Float16 v16h;
typedef __attribute__((ext_vector_type(8)))  _Float16 v8h;
typedef __attribute__((ext_vector_type(8)))  float    v8f;

#define RS_N 8192
#define RS_D 256
#define RS_EPS 1e-7f
#define RS_DIAG (-10.0f)
// log2(e)/gamma, gamma = 20
#define RS_EXP_SCALE (1.4426950408889634f / 20.0f)
// LDS B-panel row stride in halfs: 32 K-slice + 8 pad -> 20 DWORDs.
// ds_load_b128 bank starts: lane r -> (20r) mod 64 = {0,4,...,60}: conflict-free.
#define RS_BS 40

// ---- float <-> order-preserving unsigned encoding (for atomicMax on float) -
__device__ __forceinline__ unsigned int f2ord(float f) {
    unsigned int b = __float_as_uint(f);
    return (b & 0x80000000u) ? ~b : (b | 0x80000000u);
}
__device__ __forceinline__ float ord2f(unsigned int u) {
    unsigned int b = (u & 0x80000000u) ? (u & 0x7FFFFFFFu) : ~u;
    return __uint_as_float(b);
}

// ---------------------------------------------------------------------------
// Kernel 0: zero the encoded row-max array (ws is poisoned by the harness).
// ---------------------------------------------------------------------------
__global__ void rs_init_rowmax(unsigned int* rowmax_enc) {
    int i = blockIdx.x * blockDim.x + threadIdx.x;
    if (i < RS_N) rowmax_enc[i] = 0u;   // smallest ordered key; always overwritten
}

// ---------------------------------------------------------------------------
// Kernel 1: row L2-normalize feat_k / feat_g, emit f16 row-major [N][D].
// ---------------------------------------------------------------------------
__global__ __launch_bounds__(256)
void rs_normalize(const float* __restrict__ fk, const float* __restrict__ fg,
                  _Float16* __restrict__ k16, _Float16* __restrict__ g16) {
    __shared__ float sk[256];
    __shared__ float sg[256];
    const int row = blockIdx.x;
    const int tid = threadIdx.x;
    const float vk = fk[(size_t)row * RS_D + tid];
    const float vg = fg[(size_t)row * RS_D + tid];
    sk[tid] = vk * vk;
    sg[tid] = vg * vg;
    __syncthreads();
    #pragma unroll
    for (int s = 128; s > 0; s >>= 1) {
        if (tid < s) { sk[tid] += sk[tid + s]; sg[tid] += sg[tid + s]; }
        __syncthreads();
    }
    const float invk = 1.0f / (sqrtf(sk[0]) + RS_EPS);
    const float invg = 1.0f / (sqrtf(sg[0]) + RS_EPS);
    k16[(size_t)row * RS_D + tid] = (_Float16)(vk * invk);
    g16[(size_t)row * RS_D + tid] = (_Float16)(vg * invg);
}

// ---------------------------------------------------------------------------
// WMMA A fragment from global (ISA 7.12.2, wave32):
// A 16x32 f16: lane&15 = row M; lane<16 holds K {kb..+7, kb+16..+23},
//              lane>=16 holds K {kb+8..+15, kb+24..+31}.
// ---------------------------------------------------------------------------
__device__ __forceinline__ v16h rs_load_a(const _Float16* __restrict__ p,
                                          int rowBase, int kb, int lane) {
    const _Float16* base = p + (size_t)(rowBase + (lane & 15)) * RS_D + kb + ((lane >> 4) << 3);
    v8h lo = *(const v8h*)(base);
    v8h hi = *(const v8h*)(base + 16);
    v16h a;
    #pragma unroll
    for (int i = 0; i < 8; ++i) { a[i] = lo[i]; a[i + 8] = hi[i]; }
    return a;
}

// B 32x16 f16 fragment from the LDS-staged panel (row-major [col][RS_BS]):
// lane&15 = col N; lane<16 holds K {kb..+7, kb+16..+23}, lane>=16 holds K {kb+8..+15, kb+24..+31}.
__device__ __forceinline__ v16h rs_load_b_lds(const _Float16* sBmat, int ntCol, int lane) {
    const _Float16* base = sBmat + (ntCol + (lane & 15)) * RS_BS + ((lane >> 4) << 3);
    v8h lo = *(const v8h*)(base);       // ds_load_b128
    v8h hi = *(const v8h*)(base + 16);  // ds_load_b128
    v16h b;
    #pragma unroll
    for (int i = 0; i < 8; ++i) { b[i] = lo[i]; b[i + 8] = hi[i]; }
    return b;
}

// Issue one 32-K slice of both B panels into LDS via async DMA.
// Slice = 2 mats x 128 rows x 64B = 16 KB -> 4 b128 asyncs per thread.
__device__ __forceinline__ void rs_stage_async(_Float16* sBuf,
                                               const _Float16* __restrict__ k16,
                                               const _Float16* __restrict__ g16,
                                               int colBase, int kb, int tid) {
    #pragma unroll
    for (int j = 0; j < 4; ++j) {
        const int id  = tid + 256 * j;          // 0..1023
        const int mat = id >> 9;                // 0 = k, 1 = g
        const int rem = id & 511;
        const int row = rem >> 2;               // 0..127
        const int c   = rem & 3;                // 16B chunk within 64B slice row
        const _Float16* g = (mat ? g16 : k16) +
                            (size_t)(colBase + row) * RS_D + kb + c * 8;
        const unsigned l = (unsigned)(size_t)(sBuf + (mat * 128 + row) * RS_BS + c * 8);
        asm volatile("global_load_async_to_lds_b128 %0, %1, off"
                     :: "v"(l), "v"(g) : "memory");
    }
}

// ---------------------------------------------------------------------------
// Kernel 2: dual gram GEMM + logits + per-row running max.
// Grid (64,64); block = 256 threads = 8 waves; workgroup tile 128x128.
// ---------------------------------------------------------------------------
__global__ __launch_bounds__(256, 1)
void rs_gram_logits(const _Float16* __restrict__ k16,
                    const _Float16* __restrict__ g16,
                    float* __restrict__ out,
                    unsigned int* __restrict__ rowmax_enc) {
    // Double-buffered B panels: [buf][mat][128 cols][RS_BS halfs] = 40 KB.
    __shared__ __align__(16) _Float16 sB[2][2][128][RS_BS];

    const int tid     = threadIdx.x;
    const int lane    = tid & 31;
    const int wave    = tid >> 5;
    const int rowBase = blockIdx.y * 128 + wave * 16;
    const int colBase = blockIdx.x * 128;

    v8f acck[8];
    v8f accg[8];
    #pragma unroll
    for (int t = 0; t < 8; ++t) {
        #pragma unroll
        for (int p = 0; p < 8; ++p) { acck[t][p] = 0.0f; accg[t][p] = 0.0f; }
    }

    rs_stage_async(&sB[0][0][0][0], k16, g16, colBase, 0, tid);

    for (int s = 0; s < 8; ++s) {               // K = 256 in slices of 32
        const int kb = s * 32;
        if (s < 7) {
            rs_stage_async(&sB[(s + 1) & 1][0][0][0], k16, g16, colBase, kb + 32, tid);
            // in-order completion: cnt<=4 => slice s resident (next slice pending)
            asm volatile("s_wait_asynccnt 0x4" ::: "memory");
        } else {
            asm volatile("s_wait_asynccnt 0x0" ::: "memory");
        }
        __syncthreads();

        const _Float16* bk_panel = &sB[s & 1][0][0][0];
        const _Float16* bg_panel = &sB[s & 1][1][0][0];
        const v16h ak = rs_load_a(k16, rowBase, kb, lane);
        const v16h ag = rs_load_a(g16, rowBase, kb, lane);
        #pragma unroll
        for (int nt = 0; nt < 8; ++nt) {
            const v16h bk = rs_load_b_lds(bk_panel, nt * 16, lane);
            const v16h bg = rs_load_b_lds(bg_panel, nt * 16, lane);
            acck[nt] = __builtin_amdgcn_wmma_f32_16x16x32_f16(
                false, ak, false, bk, (short)0, acck[nt], false, false);
            accg[nt] = __builtin_amdgcn_wmma_f32_16x16x32_f16(
                false, ag, false, bg, (short)0, accg[nt], false, false);
        }
        __syncthreads();   // protect buffer (s&1) before it is re-staged
    }

    // C/D layout: VGPR p, lanes 0-15 -> M = p; lanes 16-31 -> M = p + 8; N = lane&15.
    const int mrow  = rowBase + ((lane >> 4) << 3);   // + p gives global row
    const int ncol0 = colBase + (lane & 15);

    float rmax[8];
    #pragma unroll
    for (int p = 0; p < 8; ++p) rmax[p] = -3.0e38f;

    #pragma unroll
    for (int nt = 0; nt < 8; ++nt) {
        const int gc = ncol0 + nt * 16;
        #pragma unroll
        for (int p = 0; p < 8; ++p) {
            const int gr = mrow + p;
            float l = acck[nt][p] / (accg[nt][p] + RS_EPS);
            if (gr == gc) l = RS_DIAG;
            out[(size_t)gr * RS_N + gc] = l;
            rmax[p] = fmaxf(rmax[p], l);
        }
    }

    // Row max across the 16 lanes of each half-wave (same row set), then atomic.
    #pragma unroll
    for (int p = 0; p < 8; ++p) {
        float r = rmax[p];
        #pragma unroll
        for (int sft = 1; sft < 16; sft <<= 1)
            r = fmaxf(r, __shfl_xor(r, sft, 32));
        if ((lane & 15) == 0)
            atomicMax(&rowmax_enc[mrow + p], f2ord(r));
    }
}

// ---------------------------------------------------------------------------
// Kernel 3: in-place  out = exp2((L - m_row) * log2(e)/20), float4 vectorized.
// ---------------------------------------------------------------------------
__global__ __launch_bounds__(256)
void rs_finalize(float* __restrict__ out,
                 const unsigned int* __restrict__ rowmax_enc) {
    const size_t i = (size_t)blockIdx.x * blockDim.x + threadIdx.x;  // float4 idx
    const int row = (int)(i >> 11);                                  // (i*4)/8192
    const float m = ord2f(rowmax_enc[row]);
    float4* o4 = (float4*)out;
    float4 v = o4[i];
    v.x = exp2f((v.x - m) * RS_EXP_SCALE);
    v.y = exp2f((v.y - m) * RS_EXP_SCALE);
    v.z = exp2f((v.z - m) * RS_EXP_SCALE);
    v.w = exp2f((v.w - m) * RS_EXP_SCALE);
    o4[i] = v;
}

// ---------------------------------------------------------------------------
// Launch: inputs = {feat_q (unused), feat_k, feat_g}, out = [1, 8192, 8192] f32.
// Workspace: k16 (4 MB) | g16 (4 MB) | rowmax (32 KB)  ~= 8.03 MB.
// ---------------------------------------------------------------------------
extern "C" void kernel_launch(void* const* d_in, const int* in_sizes, int n_in,
                              void* d_out, int out_size, void* d_ws, size_t ws_size,
                              hipStream_t stream) {
    (void)in_sizes; (void)n_in; (void)out_size; (void)ws_size;
    const float* fk = (const float*)d_in[1];
    const float* fg = (const float*)d_in[2];

    _Float16* k16 = (_Float16*)d_ws;
    _Float16* g16 = k16 + (size_t)RS_N * RS_D;
    unsigned int* rowmax_enc = (unsigned int*)(g16 + (size_t)RS_N * RS_D);
    float* out = (float*)d_out;

    rs_init_rowmax<<<(RS_N + 255) / 256, 256, 0, stream>>>(rowmax_enc);
    rs_normalize<<<RS_N, 256, 0, stream>>>(fk, fg, k16, g16);
    rs_gram_logits<<<dim3(RS_N / 128, RS_N / 128), 256, 0, stream>>>(k16, g16, out, rowmax_enc);
    rs_finalize<<<(unsigned)(((size_t)RS_N * RS_N / 4) / 256), 256, 0, stream>>>(out, rowmax_enc);
}
